// ThaiOptimizedMultiHeadAttention_74560632259193
// MI455X (gfx1250) — matrix operand order repaired and checked by reading, not compiled, not run on hardware
//
#include <hip/hip_runtime.h>

// ---------------------------------------------------------------------------
// MI455X (gfx1250) fused relative-position MHA
//   - all GEMMs via v_wmma_f32_16x16x32_bf16 (wave32, 16x16x32)
//   - attention K/V tiles staged by the Tensor Data Mover (TDM) into LDS,
//     double-buffered and synchronized with s_wait_tensorcnt
//   - flash-attention online softmax in log2 domain, per-wave 16 query rows
// ---------------------------------------------------------------------------

typedef __attribute__((ext_vector_type(16))) __bf16 v16bf;
typedef __attribute__((ext_vector_type(8)))  float  v8f;
typedef __attribute__((ext_vector_type(4)))  unsigned int v4u;
typedef __attribute__((ext_vector_type(8)))  int v8i;
typedef __attribute__((ext_vector_type(4)))  int v4i;

union FragBF { v16bf v; uint4 u[2]; };

#define SCALE_LOG2E 0.1803368801111444f  /* 0.125 * log2(e) */

#ifndef __has_builtin
#define __has_builtin(x) 0
#endif
#if __has_builtin(__builtin_amdgcn_tensor_load_to_lds) && \
    __has_builtin(__builtin_amdgcn_s_wait_tensorcnt)
#define HAVE_TDM 1
#else
#define HAVE_TDM 0
#endif

__device__ __forceinline__ unsigned short f32_bf16(float f) {
  unsigned int u = __float_as_uint(f);
  unsigned int r = (u + 0x7FFFu + ((u >> 16) & 1u)) >> 16;  // RNE
  return (unsigned short)r;
}
__device__ __forceinline__ float bf16_f32(unsigned short s) {
  return __uint_as_float(((unsigned int)s) << 16);
}
__device__ __forceinline__ unsigned int pack2(float a, float b) {
  return (unsigned int)f32_bf16(a) | ((unsigned int)f32_bf16(b) << 16);
}

#if HAVE_TDM
// 2-D tiled TDM load: 64x64 x 2-byte tile into LDS with hardware padding of
// 4 DWORDs after every 32 DWORDs stored -> matches ushort[64][72] layout.
__device__ __forceinline__ void tdm_load_2d(unsigned ldsOff, const void* g,
                                            unsigned td0, unsigned td1,
                                            unsigned stride0) {
  unsigned long long ga = (unsigned long long)(size_t)g;
  v4u g0 = {1u /*count=1*/, ldsOff, (unsigned)ga,
            (unsigned)(ga >> 32) | 0x80000000u /*type=2*/};
  v8i g1 = {
      (int)((1u << 16)    /* data_size = 2B  */
            | (1u << 20)  /* pad_enable      */
            | (4u << 22)  /* pad_interval: 32 DWORDs */
            | (3u << 25)),/* pad_amount:   4 DWORDs  */
      (int)(td0 << 16),                          /* tensor_dim0 lo */
      (int)((td0 >> 16) | (td1 << 16)),          /* dim0 hi | dim1 lo */
      (int)((td1 >> 16) | (64u << 16)),          /* dim1 hi | tile_dim0 */
      (int)64,                                   /* tile_dim1 (tile_dim2=0) */
      (int)stride0,                              /* tensor_dim0_stride lo */
      0, 0};
  v4i z4 = {0, 0, 0, 0};
#if __clang_major__ >= 23
  v8i z8 = {0, 0, 0, 0, 0, 0, 0, 0};
  __builtin_amdgcn_tensor_load_to_lds(g0, g1, z4, z4, z8, 0);
#else
  __builtin_amdgcn_tensor_load_to_lds(g0, g1, z4, z4, 0);
#endif
}
#endif

// ---------------------------------------------------------------------------
// Generic WMMA GEMM:  Out[m,n] = sum_k A[m,k] * Bw[n,k] (+ bias[n])
// Tile: 128(M) x 64(N), 8 waves, K staged 64 at a time in LDS bf16.
// OUT_MODE: 0 = f32 row-major [M,N] + bias            (final projection)
//           1 = bf16 head-split [B,H,S,64] + bias     (Q/K projections)
//           2 = bf16 head-split transposed [B,H,64,S] (V projection)
//           3 = bf16 row-major [M,N], no bias         (qr = qh @ rel^T)
// ---------------------------------------------------------------------------
template <int A_BF16, int OUT_MODE>
__global__ __launch_bounds__(256) void wmma_gemm(
    const void* __restrict__ Ap, const float* __restrict__ Bw,
    const float* __restrict__ bias, void* __restrict__ Outp,
    int M, int N, int K, int S) {
  __shared__ __align__(16) unsigned short As[128][72];  // +8 pad: bank spread
  __shared__ __align__(16) unsigned short Bs[64][72];

  const int n0 = blockIdx.x * 64;
  const int m0 = blockIdx.y * 128;
  const int tid = threadIdx.x;
  const int wave = tid >> 5, lane = tid & 31;
  const int laneN = lane & 15, laneHi = lane >> 4;

  v8f acc[4] = {};

  for (int kb = 0; kb < K; kb += 64) {
    __syncthreads();
    if (A_BF16) {
      const unsigned short* A = (const unsigned short*)Ap;
#pragma unroll
      for (int i = 0; i < 8; ++i) {
        int idx = tid + i * 256;
        int row = idx >> 4, col = (idx & 15) << 2;
        *(uint2*)&As[row][col] =
            *(const uint2*)(A + (size_t)(m0 + row) * K + kb + col);
      }
    } else {
      const float* A = (const float*)Ap;
#pragma unroll
      for (int i = 0; i < 8; ++i) {
        int idx = tid + i * 256;
        int row = idx >> 4, col = (idx & 15) << 2;
        float4 f = *(const float4*)(A + (size_t)(m0 + row) * K + kb + col);
        uint2 d; d.x = pack2(f.x, f.y); d.y = pack2(f.z, f.w);
        *(uint2*)&As[row][col] = d;
      }
    }
#pragma unroll
    for (int i = 0; i < 4; ++i) {
      int idx = tid + i * 256;
      int row = idx >> 4, col = (idx & 15) << 2;
      float4 f = *(const float4*)(Bw + (size_t)(n0 + row) * K + kb + col);
      uint2 d; d.x = pack2(f.x, f.y); d.y = pack2(f.z, f.w);
      *(uint2*)&Bs[row][col] = d;
    }
    __syncthreads();

#pragma unroll
    for (int kk = 0; kk < 64; kk += 32) {
      FragBF a;  // A 16x32 swizzle: j<8 -> K=kk+8*laneHi+j ; j>=8 -> +16
      a.u[0] = *(const uint4*)&As[wave * 16 + laneN][kk + 8 * laneHi];
      a.u[1] = *(const uint4*)&As[wave * 16 + laneN][kk + 16 + 8 * laneHi];
#pragma unroll
      for (int t = 0; t < 4; ++t) {
        FragBF b;  // B 32x16: lane col = laneN, K: j<8 -> kk+8*laneHi+j ; j>=8 -> +16
        b.u[0] = *(const uint4*)&Bs[t * 16 + laneN][kk + 8 * laneHi];
        b.u[1] = *(const uint4*)&Bs[t * 16 + laneN][kk + 16 + 8 * laneHi];
        acc[t] = __builtin_amdgcn_wmma_f32_16x16x32_bf16(
            false, a.v, false, b.v, (short)0, acc[t], false, false);
      }
    }
  }

  // epilogue: C/D layout m = r + 8*laneHi, n = laneN
#pragma unroll
  for (int t = 0; t < 4; ++t) {
    int n = n0 + t * 16 + laneN;
    float bv = (OUT_MODE == 3) ? 0.f : bias[n];
#pragma unroll
    for (int r = 0; r < 8; ++r) {
      int m = m0 + wave * 16 + r + 8 * laneHi;
      float val = acc[t][r] + bv;
      if (OUT_MODE == 0) {
        ((float*)Outp)[(size_t)m * N + n] = val;
      } else if (OUT_MODE == 3) {
        ((unsigned short*)Outp)[(size_t)m * N + n] = f32_bf16(val);
      } else {
        int b = m / S, s = m % S;
        int h = n >> 6, dk = n & 63;
        size_t off = (OUT_MODE == 1)
                         ? (((size_t)(b * 8 + h) * S) + s) * 64 + dk
                         : (((size_t)(b * 8 + h) * 64) + dk) * S + s;
        ((unsigned short*)Outp)[off] = f32_bf16(val);
      }
    }
  }
}

// ---------------------------------------------------------------------------
// Fused attention: per (b,h), 128 query rows per block (16 per wave).
// scores = (Q K^T + gather(qr, clip(i-j)+256)) * 0.125, mask, online softmax
// (log2 domain), ctx = P V (V pre-transposed). K/V tiles double-buffered via
// TDM descriptor loads, synchronized with s_wait_tensorcnt + barrier.
// ---------------------------------------------------------------------------
__global__ __launch_bounds__(256) void attn_fused(
    const unsigned short* __restrict__ qh, const unsigned short* __restrict__ kh,
    const unsigned short* __restrict__ vt, const unsigned short* __restrict__ qr,
    const int* __restrict__ mask, unsigned short* __restrict__ ctx, int S) {
  __shared__ __align__(16) unsigned short Ks[2][64][72];   // [buf][j][dk]
  __shared__ __align__(16) unsigned short Vs[2][64][72];   // [buf][dk][j]
  __shared__ __align__(16) unsigned short Ps[8][16][72];   // per-wave P tile

  const int bh = blockIdx.y;
  const int b = bh >> 3, h = bh & 7;
  const int sBase = blockIdx.x * 128;
  const int tid = threadIdx.x;
  const int wave = tid >> 5, lane = tid & 31;
  const int laneN = lane & 15, laneHi = lane >> 4;
  const int i0 = sBase + wave * 16;

  const unsigned short* khB = kh + (size_t)bh * S * 64;
  const unsigned short* vtB = vt + (size_t)bh * 64 * S;
  const unsigned short* qrB = qr + (size_t)bh * S * 512;

  // Preload this wave's Q fragments (dk = 0..63 -> two K=32 chunks).
  FragBF qf[2];
  const unsigned short* qrow = qh + ((size_t)bh * S + i0 + laneN) * 64;
  qf[0].u[0] = *(const uint4*)(qrow + 8 * laneHi);
  qf[0].u[1] = *(const uint4*)(qrow + 16 + 8 * laneHi);
  qf[1].u[0] = *(const uint4*)(qrow + 32 + 8 * laneHi);
  qf[1].u[1] = *(const uint4*)(qrow + 48 + 8 * laneHi);

  // 32-bit gather offsets: qrB is uniform -> saddr + voffset global loads.
  int rowOff[8], delta[8];
#pragma unroll
  for (int r = 0; r < 8; ++r) {
    int ig = i0 + r + 8 * laneHi;
    rowOff[r] = ig << 9;      // * 512
    delta[r] = ig + 256;
  }

  float mrow[8], lrow[8];
#pragma unroll
  for (int r = 0; r < 8; ++r) { mrow[r] = -3.0e38f; lrow[r] = 0.f; }
  v8f cacc[4] = {};

#if HAVE_TDM
  if (wave == 0) {  // prime the pipeline: tile jb=0 into buffer 0
    tdm_load_2d((unsigned)(size_t)&Ks[0][0][0], khB, 64, (unsigned)S, 64);
    tdm_load_2d((unsigned)(size_t)&Vs[0][0][0], vtB, (unsigned)S, 64,
                (unsigned)S);
  }
#endif

  for (int jb = 0; jb < S; jb += 64) {
    const int cur = (jb >> 6) & 1;
#if HAVE_TDM
    if (wave == 0) __builtin_amdgcn_s_wait_tensorcnt(0);
    __syncthreads();  // all waves: current tiles resident, old reads retired
    if (wave == 0 && jb + 64 < S) {  // prefetch next tiles into other buffer
      tdm_load_2d((unsigned)(size_t)&Ks[cur ^ 1][0][0], khB + (jb + 64) * 64,
                  64, (unsigned)S, 64);
      tdm_load_2d((unsigned)(size_t)&Vs[cur ^ 1][0][0], vtB + jb + 64,
                  (unsigned)S, 64, (unsigned)S);
    }
#else
    __syncthreads();
#pragma unroll
    for (int i = 0; i < 4; ++i) {
      int idx = tid + i * 256;
      int row = idx >> 4, col = (idx & 15) << 2;
      *(uint2*)&Ks[cur][row][col] =
          *(const uint2*)(khB + (size_t)(jb + row) * 64 + col);
      *(uint2*)&Vs[cur][row][col] =
          *(const uint2*)(vtB + (size_t)row * S + jb + col);
    }
    __syncthreads();
#endif

    // ---- scores: 4 tiles of 16 keys, K-dim = 64 (2 WMMAs each) ----
    v8f st[4] = {};
#pragma unroll
    for (int jt = 0; jt < 4; ++jt) {
#pragma unroll
      for (int kc = 0; kc < 2; ++kc) {
        FragBF kf;
        kf.u[0] =
            *(const uint4*)&Ks[cur][jt * 16 + laneN][kc * 32 + 8 * laneHi];
        kf.u[1] =
            *(const uint4*)&Ks[cur][jt * 16 + laneN][kc * 32 + 16 + 8 * laneHi];
        st[jt] = __builtin_amdgcn_wmma_f32_16x16x32_bf16(
            false, qf[kc].v, false, kf.v, (short)0, st[jt], false, false);
      }
    }

    // ---- relative bias gather + scale (log2 domain) + mask ----
#pragma unroll
    for (int jt = 0; jt < 4; ++jt) {
      int j = jb + jt * 16 + laneN;
      int mv = mask[(size_t)b * S + j];
#pragma unroll
      for (int r = 0; r < 8; ++r) {
        int idx = delta[r] - j;                       // rel + 256
        idx = idx < 0 ? 0 : (idx > 511 ? 511 : idx);  // v_med3
        float bias = bf16_f32(qrB[(unsigned)(rowOff[r] + idx)]);
        float sc = (st[jt][r] + bias) * SCALE_LOG2E;  // score * log2(e)
        st[jt][r] = mv ? sc : -1.5e9f;
      }
    }

    // ---- online softmax: row reductions across the 16-lane half ----
    float corr[8];
#pragma unroll
    for (int r = 0; r < 8; ++r) {
      float v = fmaxf(fmaxf(st[0][r], st[1][r]), fmaxf(st[2][r], st[3][r]));
#pragma unroll
      for (int off = 8; off; off >>= 1) v = fmaxf(v, __shfl_xor(v, off, 32));
      float mnew = fmaxf(mrow[r], v);
      corr[r] = exp2f(mrow[r] - mnew);
      mrow[r] = mnew;
    }
    float rsum[8];
#pragma unroll
    for (int r = 0; r < 8; ++r) rsum[r] = 0.f;
#pragma unroll
    for (int jt = 0; jt < 4; ++jt) {
#pragma unroll
      for (int r = 0; r < 8; ++r) {
        float p = exp2f(st[jt][r] - mrow[r]);
        rsum[r] += p;
        Ps[wave][r + 8 * laneHi][jt * 16 + laneN] = f32_bf16(p);  // C->A relay
      }
    }
#pragma unroll
    for (int r = 0; r < 8; ++r) {
      float v = rsum[r];
#pragma unroll
      for (int off = 8; off; off >>= 1) v += __shfl_xor(v, off, 32);
      lrow[r] = lrow[r] * corr[r] + v;
#pragma unroll
      for (int t = 0; t < 4; ++t) cacc[t][r] = cacc[t][r] * corr[r];
    }

    // ---- ctx += P @ V (per-wave private P tile; DS ops in-order) ----
#pragma unroll
    for (int kc = 0; kc < 2; ++kc) {
      FragBF pf;
      pf.u[0] = *(const uint4*)&Ps[wave][laneN][kc * 32 + 8 * laneHi];
      pf.u[1] = *(const uint4*)&Ps[wave][laneN][kc * 32 + 16 + 8 * laneHi];
#pragma unroll
      for (int t = 0; t < 4; ++t) {
        FragBF vf;
        vf.u[0] =
            *(const uint4*)&Vs[cur][t * 16 + laneN][kc * 32 + 8 * laneHi];
        vf.u[1] =
            *(const uint4*)&Vs[cur][t * 16 + laneN][kc * 32 + 16 + 8 * laneHi];
        cacc[t] = __builtin_amdgcn_wmma_f32_16x16x32_bf16(
            false, pf.v, false, vf.v, (short)0, cacc[t], false, false);
      }
    }
  }

  // ---- normalize, store ctx bf16 [B,S,512] (col = h*64 + dk) ----
#pragma unroll
  for (int r = 0; r < 8; ++r) {
    float inv = 1.f / lrow[r];
    int s_g = i0 + r + 8 * laneHi;
#pragma unroll
    for (int t = 0; t < 4; ++t) {
      ctx[((size_t)b * S + s_g) * 512 + h * 64 + t * 16 + laneN] =
          f32_bf16(cacc[t][r] * inv);
    }
  }
}

// ---------------------------------------------------------------------------
extern "C" void kernel_launch(void* const* d_in, const int* in_sizes, int n_in,
                              void* d_out, int out_size, void* d_ws, size_t ws_size,
                              hipStream_t stream) {
  const int Bb = 2, S = 2048, D = 512, DK = 64;
  const int M = Bb * S;    // 4096 token rows
  const int BH = Bb * 8;   // 16 (batch*heads)

  const float* q  = (const float*)d_in[0];
  const float* k  = (const float*)d_in[1];
  const float* v  = (const float*)d_in[2];
  const int*  msk = (const int*)d_in[3];
  const float* wq = (const float*)d_in[4];
  const float* bq = (const float*)d_in[5];
  const float* wk = (const float*)d_in[6];
  const float* bk = (const float*)d_in[7];
  const float* wv = (const float*)d_in[8];
  const float* bv = (const float*)d_in[9];
  const float* wo = (const float*)d_in[10];
  const float* bo = (const float*)d_in[11];
  const float* rel = (const float*)d_in[12];

  char* ws = (char*)d_ws;
  unsigned short* qh  = (unsigned short*)(ws);                            // 4 MB
  unsigned short* khb = (unsigned short*)(ws + (size_t)4  * 1024 * 1024); // 4 MB
  unsigned short* vtb = (unsigned short*)(ws + (size_t)8  * 1024 * 1024); // 4 MB
  unsigned short* ctx = (unsigned short*)(ws + (size_t)12 * 1024 * 1024); // 4 MB
  unsigned short* qrb = (unsigned short*)(ws + (size_t)16 * 1024 * 1024); // 32 MB

  dim3 blk(256);
  dim3 gproj(D / 64, M / 128);  // (8, 32)
  wmma_gemm<0, 1><<<gproj, blk, 0, stream>>>(q, wq, bq, qh,  M, D, D, S);
  wmma_gemm<0, 1><<<gproj, blk, 0, stream>>>(k, wk, bk, khb, M, D, D, S);
  wmma_gemm<0, 2><<<gproj, blk, 0, stream>>>(v, wv, bv, vtb, M, D, D, S);

  dim3 gqr(512 / 64, (BH * S) / 128);  // (8, 256): qr = qh @ rel^T, K = 64
  wmma_gemm<1, 3><<<gqr, blk, 0, stream>>>(qh, rel, nullptr, qrb, BH * S, 512, DK, S);

  dim3 gattn(S / 128, BH);  // (16, 16)
  attn_fused<<<gattn, blk, 0, stream>>>(qh, khb, vtb, qrb, msk, ctx, S);

  wmma_gemm<1, 0><<<gproj, blk, 0, stream>>>(ctx, wo, bo, (float*)d_out, M, D, D, S);
}
